// DeformConv2d_44573170598788
// MI455X (gfx1250) — matrix operand order repaired and checked by reading, not compiled, not run on hardware
//
#include <hip/hip_runtime.h>
#include <hip/hip_bf16.h>

// ---------------------------------------------------------------------------
// DCNv2 (modulated deformable conv) for gfx1250 / MI455X.
//   B=8, C=128, H=W=64, Co=128, K=3, dil=2, pad=2, stride=1, DG=2 -> Ho=Wo=64
// Pipeline:
//   1) offset conv (grouped, 54ch) fused into py/px/mask        [VALU]
//   2) weight -> f16, permuted to K-order (g, kk, c)            [VALU]
//   3) per-batch modulated bilinear gather -> f16 im2col colT   [VMEM]
//   4) per-batch GEMM out[128,4096] = Wh[128,1152] x colT^T     [v_wmma]
// ---------------------------------------------------------------------------

#define BB   8
#define CC   128
#define HH   64
#define WW   64
#define CO   128
#define KK   9
#define DG   2
#define CG   64          // C / DG
#define NN   4096        // Ho*Wo
#define KD   1152        // DG*CG*KK  (GEMM reduction dim)
#define OCH  54          // DG*3*KK offset-conv output channels

typedef __attribute__((ext_vector_type(16))) _Float16 v16h;
typedef __attribute__((ext_vector_type(8)))  _Float16 v8h;
typedef __attribute__((ext_vector_type(8)))  float    v8f;

// ---------------------------------------------------------------------------
// Stage 1: grouped dilated 3x3 conv (54 channels) fused with coordinate /
// mask generation.  One thread per (b, ch, ho, wo).
// ---------------------------------------------------------------------------
__global__ void dcn_offset_coords(const float* __restrict__ x,
                                  const float* __restrict__ ow,
                                  const float* __restrict__ ob,
                                  float* __restrict__ py,
                                  float* __restrict__ px,
                                  float* __restrict__ mk)
{
    int tid = blockIdx.x * blockDim.x + threadIdx.x;
    const int total = BB * OCH * NN;
    if (tid >= total) return;

    int n  = tid & (NN - 1);
    int t  = tid >> 12;
    int ch = t % OCH;
    int b  = t / OCH;
    int ho = n >> 6;
    int wo = n & 63;

    int ic0 = (ch / 27) * CG;          // conv group of this output channel
    float acc = ob[ch];
    const float* wp = ow + ch * CG * KK;
    const float* xp = x + (size_t)(b * CC + ic0) * (HH * WW);

    for (int c = 0; c < CG; ++c) {
        const float* xc = xp + c * (HH * WW);
        const float* wc = wp + c * KK;
#pragma unroll
        for (int ky = 0; ky < 3; ++ky) {
            int iy  = ho - 2 + ky * 2;                       // pad=2, dil=2
            int iyc = min(max(iy, 0), HH - 1);
            bool yok = (iy >= 0) && (iy < HH);
#pragma unroll
            for (int kx = 0; kx < 3; ++kx) {
                int ix  = wo - 2 + kx * 2;
                int ixc = min(max(ix, 0), WW - 1);
                float v = xc[iyc * WW + ixc];
                float m = (yok && ix >= 0 && ix < WW) ? 1.0f : 0.0f;
                acc = fmaf(v * m, wc[ky * 3 + kx], acc);
            }
        }
    }

    if (ch < 36) {                                           // offsets
        int pair = ch >> 1, d = ch & 1;
        int g = pair / KK, kk = pair % KK;
        int ky = kk / 3, kx = kk % 3;
        int idx = ((b * DG + g) * KK + kk) * NN + n;
        if (d == 0) py[idx] = acc + (float)(ky * 2 + ho - 2);
        else        px[idx] = acc + (float)(kx * 2 + wo - 2);
    } else {                                                 // masks (2*sigmoid)
        int m = ch - 36;
        int g = m / KK, kk = m % KK;
        int idx = ((b * DG + g) * KK + kk) * NN + n;
        mk[idx] = 2.0f / (1.0f + __expf(-acc));
    }
}

// ---------------------------------------------------------------------------
// Stage 2: weight f32 [Co, C, 3, 3] -> f16 [Co][g*576 + kk*64 + c]
// (K-order chosen so the gather writes channel-contiguous halves).
// ---------------------------------------------------------------------------
__global__ void dcn_weight_half(const float* __restrict__ w,
                                _Float16* __restrict__ wh)
{
    int tid = blockIdx.x * blockDim.x + threadIdx.x;
    const int total = CO * KD;
    if (tid >= total) return;
    int kidx = tid % KD;
    int o    = tid / KD;
    int g  = kidx / (KK * CG);
    int r  = kidx % (KK * CG);
    int kk = r / CG;
    int c  = r % CG;
    wh[(size_t)o * KD + kidx] = (_Float16)w[((size_t)(o * CC + g * CG + c)) * KK + kk];
}

// ---------------------------------------------------------------------------
// Stage 3 (per batch): modulated bilinear gather -> f16 colT[n][kidx].
// One wave per (g, kk, n); lanes cover the 64 channels (2 each).
// Zero padding folded into the corner weights; clamped reads stay in-bounds.
// ---------------------------------------------------------------------------
__global__ void dcn_gather(const float* __restrict__ x,
                           const float* __restrict__ py,
                           const float* __restrict__ px,
                           const float* __restrict__ mk,
                           _Float16* __restrict__ col, int b)
{
    int wid  = (blockIdx.x * blockDim.x + threadIdx.x) >> 5;
    int lane = threadIdx.x & 31;
    const int S = DG * KK * NN;
    if (wid >= S) return;

    int g  = wid / (KK * NN);
    int r  = wid % (KK * NN);
    int kk = r >> 12;
    int n  = r & (NN - 1);

    int sidx = ((b * DG + g) * KK + kk) * NN + n;
    float Y = py[sidx], X = px[sidx], M = mk[sidx];

    float y0f = floorf(Y), x0f = floorf(X);
    int   y0  = (int)y0f,  x0  = (int)x0f;
    float wy  = Y - y0f,   wx  = X - x0f;

    bool y0ok = (y0 >= 0)     && (y0 < HH);
    bool y1ok = (y0 + 1 >= 0) && (y0 + 1 < HH);
    bool x0ok = (x0 >= 0)     && (x0 < WW);
    bool x1ok = (x0 + 1 >= 0) && (x0 + 1 < WW);

    float w00 = (1.f - wy) * (1.f - wx) * ((y0ok && x0ok) ? 1.f : 0.f);
    float w01 = (1.f - wy) * wx         * ((y0ok && x1ok) ? 1.f : 0.f);
    float w10 = wy * (1.f - wx)         * ((y1ok && x0ok) ? 1.f : 0.f);
    float w11 = wy * wx                 * ((y1ok && x1ok) ? 1.f : 0.f);

    int y0c = min(max(y0,     0), HH - 1);
    int y1c = min(max(y0 + 1, 0), HH - 1);
    int x0c = min(max(x0,     0), WW - 1);
    int x1c = min(max(x0 + 1, 0), WW - 1);

    int i00 = y0c * WW + x0c, i01 = y0c * WW + x1c;
    int i10 = y1c * WW + x0c, i11 = y1c * WW + x1c;

    const float* xb = x + (size_t)(b * CC + g * CG) * (HH * WW);
    _Float16* cp = col + (size_t)n * KD + g * (KK * CG) + kk * CG;

#pragma unroll
    for (int h = 0; h < 2; ++h) {
        int c = lane + h * 32;
        const float* xc = xb + c * (HH * WW);
        float v = w00 * xc[i00] + w01 * xc[i01] + w10 * xc[i10] + w11 * xc[i11];
        cp[c] = (_Float16)(v * M);
    }
}

// ---------------------------------------------------------------------------
// Stage 4 (per batch): out[128, 4096] = Wh[128, 1152] x colT^T via
// v_wmma_f32_16x16x32_f16.  Block = 256 threads = 8 waves; wave w owns output
// rows [16w, 16w+16) x columns [64*blockIdx.x, +64) -> 4 accumulators with
// A-fragment reuse.  K = 1152 -> 36 WMMA steps, no remainder.
// ---------------------------------------------------------------------------
__global__ void dcn_gemm(const _Float16* __restrict__ wh,
                         const _Float16* __restrict__ col,
                         float* __restrict__ out, int b)
{
    const int lane = threadIdx.x & 31;
    const int wave = threadIdx.x >> 5;        // 0..7 -> M strip
    const int row  = lane & 15;
    const int hi   = lane >> 4;               // half-wave select
    const int m0   = wave * 16;
    const int n0   = blockIdx.x * 64;

    // A operand, contiguous k order used here: lanes 0-15 hold K=0..15, lanes 16-31 hold K=16..31.
    const _Float16* arow = wh + (size_t)(m0 + row) * KD + 16 * hi;
    // B operand, same contiguous k order as A: lanes 0-15 hold K=0..15, lanes 16-31 hold K=16..31.
    const _Float16* b0p = col + (size_t)(n0 +  0 + row) * KD + 16 * hi;
    const _Float16* b1p = col + (size_t)(n0 + 16 + row) * KD + 16 * hi;
    const _Float16* b2p = col + (size_t)(n0 + 32 + row) * KD + 16 * hi;
    const _Float16* b3p = col + (size_t)(n0 + 48 + row) * KD + 16 * hi;

    v8f acc0 = {}, acc1 = {}, acc2 = {}, acc3 = {};

#pragma unroll 2
    for (int k0 = 0; k0 < KD; k0 += 32) {
        v8h alo = *(const v8h*)(arow + k0);
        v8h ahi = *(const v8h*)(arow + k0 + 8);
        v16h a = __builtin_shufflevector(alo, ahi,
                 0,1,2,3,4,5,6,7,8,9,10,11,12,13,14,15);

        v8h t0 = *(const v8h*)(b0p + k0);
        v8h t1 = *(const v8h*)(b0p + k0 + 8);
        v16h bb = __builtin_shufflevector(t0, t1,
                  0,1,2,3,4,5,6,7,8,9,10,11,12,13,14,15);
        acc0 = __builtin_amdgcn_wmma_f32_16x16x32_f16(false, a, false, bb,
                                                      (short)0, acc0, false, false);

        t0 = *(const v8h*)(b1p + k0);
        t1 = *(const v8h*)(b1p + k0 + 8);
        bb = __builtin_shufflevector(t0, t1,
             0,1,2,3,4,5,6,7,8,9,10,11,12,13,14,15);
        acc1 = __builtin_amdgcn_wmma_f32_16x16x32_f16(false, a, false, bb,
                                                      (short)0, acc1, false, false);

        t0 = *(const v8h*)(b2p + k0);
        t1 = *(const v8h*)(b2p + k0 + 8);
        bb = __builtin_shufflevector(t0, t1,
             0,1,2,3,4,5,6,7,8,9,10,11,12,13,14,15);
        acc2 = __builtin_amdgcn_wmma_f32_16x16x32_f16(false, a, false, bb,
                                                      (short)0, acc2, false, false);

        t0 = *(const v8h*)(b3p + k0);
        t1 = *(const v8h*)(b3p + k0 + 8);
        bb = __builtin_shufflevector(t0, t1,
             0,1,2,3,4,5,6,7,8,9,10,11,12,13,14,15);
        acc3 = __builtin_amdgcn_wmma_f32_16x16x32_f16(false, a, false, bb,
                                                      (short)0, acc3, false, false);
    }

    // C/D layout: element r of lane l -> M = r + 8*(l>=16), N = l%16.
    float* op = out + (size_t)b * CO * NN;
#pragma unroll
    for (int r = 0; r < 8; ++r) {
        int m = m0 + r + 8 * hi;
        op[(size_t)m * NN + n0 +  0 + row] = acc0[r];
        op[(size_t)m * NN + n0 + 16 + row] = acc1[r];
        op[(size_t)m * NN + n0 + 32 + row] = acc2[r];
        op[(size_t)m * NN + n0 + 48 + row] = acc3[r];
    }
}

// ---------------------------------------------------------------------------
// Workspace layout (bytes):
//   py   : 0         .. 2359296   (B*DG*KK*N f32)
//   px   : 2359296   .. 4718592
//   mask : 4718592   .. 7077888
//   wh   : 7077888   .. 7372800   (Co*Kd f16)
//   colT : 7372800   .. 16809984  (N*Kd f16, reused per batch)
// ---------------------------------------------------------------------------
extern "C" void kernel_launch(void* const* d_in, const int* in_sizes, int n_in,
                              void* d_out, int out_size, void* d_ws, size_t ws_size,
                              hipStream_t stream) {
    (void)in_sizes; (void)n_in; (void)out_size; (void)ws_size;

    const float* x  = (const float*)d_in[0];
    const float* ow = (const float*)d_in[1];
    const float* ob = (const float*)d_in[2];
    const float* w  = (const float*)d_in[3];
    float* out = (float*)d_out;

    char* ws = (char*)d_ws;
    float*    py  = (float*)(ws + 0);
    float*    px  = (float*)(ws + 2359296);
    float*    mk  = (float*)(ws + 4718592);
    _Float16* wh  = (_Float16*)(ws + 7077888);
    _Float16* col = (_Float16*)(ws + 7372800);

    {   // Stage 1: offset conv + coordinate/mask generation (all batches)
        int total = BB * OCH * NN;
        dcn_offset_coords<<<(total + 255) / 256, 256, 0, stream>>>(x, ow, ob, py, px, mk);
    }
    {   // Stage 2: weight -> f16, GEMM K-order
        int total = CO * KD;
        dcn_weight_half<<<(total + 255) / 256, 256, 0, stream>>>(w, wh);
    }
    for (int b = 0; b < BB; ++b) {
        // Stage 3: gather -> colT (one wave per sample, 8 waves/block)
        dcn_gather<<<(DG * KK * NN) / 8, 256, 0, stream>>>(x, py, px, mk, col, b);
        // Stage 4: WMMA GEMM (grid 64 x block 256 covers 128 x 4096 outputs)
        dcn_gemm<<<NN / 64, 256, 0, stream>>>(wh, col, out, b);
    }
}
